// VectorQuantizer_62612033241626
// MI455X (gfx1250) — compile-verified, matrix operand order checked by reading, not run on hardware
//
#include <hip/hip_runtime.h>
#include <hip/hip_bf16.h>

// ---------------- problem constants ----------------
#define EDIM    256                // embedding dim (GEMM reduction dim)
#define KCODES  8192               // codebook size
#define NROWS   32768              // B*L tokens
#define MBLK    128                // rows per block (8 waves x 16)
#define NTILES  (KCODES / 16)      // 512 column tiles of 16

// d_out layout (flat, return order): quantized[8388608] | dl | cl | idx[32768]
#define LOSS_OFF (NROWS * EDIM)
#define IDX_OFF  (LOSS_OFF + 2)

// workspace layout
#define WS_NORM_OFF 0
#define WS_IDX_OFF  (KCODES * 4)            // 32 KB
#define WS_ET_OFF   (WS_IDX_OFF + NROWS * 4) // +128 KB -> e_t bf16 [K][E] = 4 MB

typedef __bf16 v16bf __attribute__((ext_vector_type(16)));
typedef __bf16 v8bf  __attribute__((ext_vector_type(8)));
typedef float  v8f   __attribute__((ext_vector_type(8)));

union bfcv { unsigned short u; __bf16 b; };

static __device__ inline __bf16 f2bf(float f) {
    // round-to-nearest-even f32 -> bf16
    unsigned u = __float_as_uint(f);
    unsigned r = (u + 0x7FFFu + ((u >> 16) & 1u)) >> 16;
    bfcv c; c.u = (unsigned short)r;
    return c.b;
}

// One 16-bit WMMA A/B fragment slice for this lane: two contiguous 16B chunks
// (ISA 16-bit layout: per-lane K chunks at off0 and off0+16).
static __device__ inline v16bf load_frag_bf16(const __bf16* base, int off0) {
    v8bf lo = *(const v8bf*)(base + off0);
    v8bf hi = *(const v8bf*)(base + off0 + 16);
    return __builtin_shufflevector(lo, hi, 0,1,2,3,4,5,6,7,8,9,10,11,12,13,14,15);
}

// Same slice but from f32 source, converted to bf16 in registers.
static __device__ inline v16bf load_frag_f32(const float* base, int off0) {
    float4 f0 = *(const float4*)(base + off0);
    float4 f1 = *(const float4*)(base + off0 + 4);
    float4 f2 = *(const float4*)(base + off0 + 16);
    float4 f3 = *(const float4*)(base + off0 + 20);
    v16bf a;
    a[0]=f2bf(f0.x); a[1]=f2bf(f0.y); a[2]=f2bf(f0.z); a[3]=f2bf(f0.w);
    a[4]=f2bf(f1.x); a[5]=f2bf(f1.y); a[6]=f2bf(f1.z); a[7]=f2bf(f1.w);
    a[8]=f2bf(f2.x); a[9]=f2bf(f2.y); a[10]=f2bf(f2.z); a[11]=f2bf(f2.w);
    a[12]=f2bf(f3.x); a[13]=f2bf(f3.y); a[14]=f2bf(f3.z); a[15]=f2bf(f3.w);
    return a;
}

// ---- kernel A: codebook norms + one-time f32->bf16 transpose (K-major) ----
__global__ void vq_prep_kernel(const float* __restrict__ e, float* __restrict__ wsNorm,
                               __bf16* __restrict__ et, float* __restrict__ out) {
    int k = blockIdx.x * 256 + threadIdx.x;           // 0..8191
    float s = 0.f;
    for (int e0 = 0; e0 < EDIM; e0 += 8) {
        v8bf pk;
        #pragma unroll
        for (int j = 0; j < 8; ++j) {
            float v = e[(size_t)(e0 + j) * KCODES + k];  // coalesced across k
            s = fmaf(v, v, s);
            pk[j] = f2bf(v);
        }
        *(v8bf*)(et + (size_t)k * EDIM + e0) = pk;       // 16B store, column k row
    }
    wsNorm[k] = s;
    if (k < 2) out[LOSS_OFF + k] = 0.f;               // re-zero every call (graph replay)
}

// ---- kernel B: fused bf16-WMMA GEMM + argmin, no LDS, no barriers ----
__global__ void vq_argmin_kernel(const float* __restrict__ x, const __bf16* __restrict__ et,
                                 const float* __restrict__ wsNorm, int* __restrict__ wsIdx,
                                 float* __restrict__ out) {
    const int tid  = threadIdx.x;                     // 256 threads = 8 waves
    const int lane = tid & 31;
    const int w    = tid >> 5;                        // wave -> 16-row strip
    const int half = lane >> 4;
    const int l15  = lane & 15;
    const int rowBase = blockIdx.x * MBLK;

    // A fragments: this wave's 16 rows of x, all 8 K-steps, held in regs for
    // the entire loop (64 VGPRs). Converted f32->bf16 once.
    const float* xr = x + (size_t)(rowBase + w * 16 + l15) * EDIM;
    v16bf af[8];
    #pragma unroll
    for (int s = 0; s < 8; ++s)
        af[s] = load_frag_f32(xr, s * 32 + half * 8);

    float bestd[8];
    int   besti[8];
    #pragma unroll
    for (int j = 0; j < 8; ++j) { bestd[j] = 3.0e38f; besti[j] = 0; }

    #pragma unroll 1
    for (int tt = 0; tt < NTILES; ++tt) {
        const int col = tt * 16 + l15;                // this lane's N column
        const __bf16* er = et + (size_t)col * EDIM;   // contiguous bf16 column

        if (tt + 8 < NTILES)                          // global_prefetch_b8, 8 tiles ahead
            __builtin_prefetch(er + 8 * 16 * EDIM, 0, 1);

        // preload all 8 B fragments so loads overlap the WMMA chain
        v16bf bf[8];
        #pragma unroll
        for (int s = 0; s < 8; ++s)
            bf[s] = load_frag_bf16(er, s * 32 + half * 8);

        v8f acc = {};
        #pragma unroll
        for (int s = 0; s < 8; ++s)
            acc = __builtin_amdgcn_wmma_f32_16x16x32_bf16(
                    false, af[s], false, bf[s], (short)0, acc, false, false);

        const float en = wsNorm[col];
        #pragma unroll
        for (int j = 0; j < 8; ++j) {
            float d = fmaf(-2.0f, acc[j], en);        // ||e||^2 - 2 x.e
            if (d < bestd[j]) { bestd[j] = d; besti[j] = col; }
        }
    }

    // reduce argmin across the 16 lanes sharing each row (per half-wave)
    #pragma unroll
    for (int j = 0; j < 8; ++j) {
        #pragma unroll
        for (int m = 8; m >= 1; m >>= 1) {
            float od = __shfl_xor(bestd[j], m, 32);
            int   oi = __shfl_xor(besti[j], m, 32);
            if (od < bestd[j] || (od == bestd[j] && oi < besti[j])) {
                bestd[j] = od; besti[j] = oi;
            }
        }
        if (l15 == 0) {                               // lanes 0 and 16 hold results
            int rowM = j + (half ? 8 : 0);            // C layout: M = vgpr (+8 hi half)
            int row  = rowBase + w * 16 + rowM;
            wsIdx[row] = besti[j];
            out[IDX_OFF + row] = (float)besti[j];
        }
    }
}

// ---- kernel C: gather winning codes, quantized out, MSE losses ----
__global__ void vq_gather_kernel(const float* __restrict__ x, const float* __restrict__ e,
                                 const int* __restrict__ wsIdx, float* __restrict__ out) {
    __shared__ float red[256];
    const int row = blockIdx.x;
    const int t   = threadIdx.x;
    const int k   = wsIdx[row];
    float q  = e[(size_t)t * KCODES + k];             // f32 column gather (L2-resident)
    float xv = x[(size_t)row * EDIM + t];
    out[(size_t)row * EDIM + t] = q;                  // straight-through == q
    float d = xv - q;
    red[t] = d * d;
    __syncthreads();
    #pragma unroll
    for (int s = 128; s > 0; s >>= 1) {
        if (t < s) red[t] += red[t + s];
        __syncthreads();
    }
    if (t == 0) {
        float v = red[0] * (1.0f / (float)(NROWS * EDIM));
        atomicAdd(out + LOSS_OFF + 0, v);             // dictionary_loss
        atomicAdd(out + LOSS_OFF + 1, v);             // commitment_loss (identical)
    }
}

extern "C" void kernel_launch(void* const* d_in, const int* in_sizes, int n_in,
                              void* d_out, int out_size, void* d_ws, size_t ws_size,
                              hipStream_t stream) {
    const float* x = (const float*)d_in[0];           // [32,1024,256] f32
    const float* e = (const float*)d_in[1];           // [256,8192]    f32
    float* out     = (float*)d_out;

    float*  wsNorm = (float*)((char*)d_ws + WS_NORM_OFF);  // 8192 f32
    int*    wsIdx  = (int*)((char*)d_ws + WS_IDX_OFF);     // 32768 i32
    __bf16* et     = (__bf16*)((char*)d_ws + WS_ET_OFF);   // [K][E] bf16, 4 MB

    vq_prep_kernel<<<KCODES / 256, 256, 0, stream>>>(e, wsNorm, et, out);
    vq_argmin_kernel<<<NROWS / MBLK, 256, 0, stream>>>(x, et, wsNorm, wsIdx, out);
    vq_gather_kernel<<<NROWS, 256, 0, stream>>>(x, e, wsIdx, out);
    (void)in_sizes; (void)n_in; (void)out_size; (void)ws_size;
}
